// PhedVec_11793980195385
// MI455X (gfx1250) — matrix operand order repaired and checked by reading, not compiled, not run on hardware
//
#include <hip/hip_runtime.h>
#include <math.h>

// ---------------------------------------------------------------------------
// PhedVec loss on gfx1250: bf16 WMMA Gram with TDM-staged LDS double buffering
// ---------------------------------------------------------------------------

typedef __bf16 bf16_t;
typedef __attribute__((ext_vector_type(16))) __bf16 v16bf;
typedef __attribute__((ext_vector_type(8)))  __bf16 v8bf;
typedef __attribute__((ext_vector_type(8)))  float  v8f;
typedef __attribute__((ext_vector_type(4)))  unsigned int v4u;
typedef __attribute__((ext_vector_type(8)))  int v8i;
typedef __attribute__((ext_vector_type(4)))  int v4i;

#define V_SZ 12288
#define D_SZ 128
#define B_SZ 256
#define L_SZ 64
#define P_SZ 200000
#define C_SZ 1500

// Gram tiling: 4 waves/block, 64 M rows per block, N sweep split in 2.
#define GRAM_WAVES   4
#define GRAM_MROWS   64
#define GRAM_NSPLIT  2
#define STAGE_COLS   64                      // E rows staged per buffer
#define STAGE_BYTES  (STAGE_COLS * D_SZ * 2) // 16384 B
#define NSTAGE       (V_SZ / GRAM_NSPLIT / STAGE_COLS)  // 96

#if __has_builtin(__builtin_amdgcn_tensor_load_to_lds)
#define HAVE_TDM 1
#else
#define HAVE_TDM 0
#endif

union AV { v16bf v; v8bf h[2]; };

// ---------------- init: zero norms + the two loss accumulators --------------
__global__ void __launch_bounds__(256) init_k(float* norms, float* accs) {
  int i = blockIdx.x * blockDim.x + threadIdx.x;
  if (i < V_SZ) norms[i] = 0.0f;
  if (i == 0) { accs[0] = 0.0f; accs[1] = 0.0f; }
}

// ---------------- fp32 -> bf16 copy of the embedding ------------------------
__global__ void cvt_bf16_k(const float* __restrict__ in,
                           unsigned short* __restrict__ out, int n) {
  int i = blockIdx.x * blockDim.x + threadIdx.x;
  if (i < n) {
    bf16_t b = (bf16_t)in[i];
    out[i] = __builtin_bit_cast(unsigned short, b);
  }
}

#if HAVE_TDM
// Issue a TDM copy of STAGE_BYTES contiguous bytes (global -> LDS).
// D#: 1-D tile, data_size=8B, tile_dim0 = STAGE_BYTES/8 = 2048 elements.
// This toolchain exposes the 6-argument builtin form:
//   (uint32x4 g0, int32x8 g1, int32x4 g2, int32x4 g3, int32x8 extra, i32 cpol)
__device__ __forceinline__ void tdm_stage_load(const void* gsrc,
                                               unsigned lds_off) {
  unsigned long long ga = (unsigned long long)(uintptr_t)gsrc;
  v4u g0;
  g0[0] = 1u;                                   // count=1, user mode
  g0[1] = lds_off;                              // lds_addr [63:32]
  g0[2] = (unsigned)(ga & 0xFFFFFFFFu);         // global_addr low
  g0[3] = (unsigned)((ga >> 32) & 0x01FFFFFFu)  // global_addr high (57-bit)
          | 0x80000000u;                        // type=2 ("image") [127:126]
  v8i g1;
  g1[0] = 0x00030000;            // workgroup_mask=0, data_size=3 (8 bytes)
  g1[1] = (int)(2048u << 16);    // tensor_dim0[15:0] @ [63:48]
  g1[2] = (int)(1u << 16);       // tensor_dim0 hi=0, tensor_dim1=1
  g1[3] = (int)(2048u << 16);    // tile_dim0=2048 @ [127:112]
  g1[4] = 0;                     // tile_dim1=0, tile_dim2=0 (unused)
  g1[5] = 2048;                  // tensor_dim0_stride low 32
  g1[6] = 0;                     // stride hi, tensor_dim1_stride low
  g1[7] = 0;
  v4i z4 = {0, 0, 0, 0};
  v8i z8 = {0, 0, 0, 0, 0, 0, 0, 0};
  __builtin_amdgcn_tensor_load_to_lds(g0, g1, z4, z4, z8, 0);
}
#endif

// ---------------- Gram: norms[m] += sum_n exp(E[m] . E[n]) ------------------
// Block: 4 waves, each owning a 16-row M tile (64 rows/block), sweeping half
// of the N range. B panels (64 E-rows = 16 KB) are staged into double-
// buffered LDS by the Tensor Data Mover and reused by all 4 waves.
__global__ void __launch_bounds__(128)
gram_norms_k(const unsigned short* __restrict__ EbRaw,
             float* __restrict__ norms) {
  __shared__ __align__(128) unsigned char lds_buf[2][STAGE_BYTES];
  const bf16_t* Eb = reinterpret_cast<const bf16_t*>(EbRaw);

  const int lane  = threadIdx.x & 31;
  const int wv    = threadIdx.x >> 5;             // 0..3
  const int mgrp  = blockIdx.x >> 1;
  const int nhalf = blockIdx.x & 1;
  const int m0    = mgrp * GRAM_MROWS + wv * 16;  // this wave's 16 rows
  const int nbase = nhalf * (V_SZ / GRAM_NSPLIT);
  const int mrow  = lane & 15;
  const int half  = lane >> 4;

  // --- A: 16x128 bf16 panel resident in registers (contiguous k = 16*half + s) -----
  AV a[4];
  const bf16_t* arow = Eb + (size_t)(m0 + mrow) * D_SZ;
#pragma unroll
  for (int c = 0; c < 4; ++c) {
    a[c].h[0] = *(const v8bf*)(arow + c * 32 + half * 16);
    a[c].h[1] = *(const v8bf*)(arow + c * 32 + half * 16 + 8);
  }

  v8f acc = {0.f, 0.f, 0.f, 0.f, 0.f, 0.f, 0.f, 0.f};

#if HAVE_TDM
  if (threadIdx.x < 32) {  // wave 0 issues stage 0 (TDM ignores EXEC)
    tdm_stage_load(Eb + (size_t)nbase * D_SZ,
                   (unsigned)(uintptr_t)&lds_buf[0][0]);
  }
#endif

  for (int s = 0; s < NSTAGE; ++s) {
    const int buf = s & 1;
#if HAVE_TDM
    if (threadIdx.x < 32 && s + 1 < NSTAGE) {
      tdm_stage_load(Eb + (size_t)(nbase + (s + 1) * STAGE_COLS) * D_SZ,
                     (unsigned)(uintptr_t)&lds_buf[buf ^ 1][0]);
    }
    if (s + 1 < NSTAGE) __builtin_amdgcn_s_wait_tensorcnt(1);
    else                __builtin_amdgcn_s_wait_tensorcnt(0);
    __syncthreads();                 // staged panel visible to all waves
#else
    {  // fallback: cooperative synchronous copy (128 B per thread)
      const uint4* src = (const uint4*)(Eb + (size_t)(nbase + s * STAGE_COLS) * D_SZ);
      uint4* dst = (uint4*)&lds_buf[buf][0];
#pragma unroll
      for (int q = 0; q < 8; ++q) {
        int idx = threadIdx.x + q * 128;         // 1024 uint4 total
        dst[idx] = src[idx];
      }
      __syncthreads();
    }
#endif

    // --- compute 4 N-tiles from the staged panel (B read via ds_load) -------
    const bf16_t* bpanel = (const bf16_t*)&lds_buf[buf][0];
#pragma unroll
    for (int t = 0; t < 4; ++t) {
      const bf16_t* brow = bpanel + (size_t)(t * 16 + mrow) * D_SZ + half * 16;
      v16bf b0 = *(const v16bf*)(brow);
      v16bf b1 = *(const v16bf*)(brow + 32);
      v16bf b2 = *(const v16bf*)(brow + 64);
      v16bf b3 = *(const v16bf*)(brow + 96);

      v8f c = {0.f, 0.f, 0.f, 0.f, 0.f, 0.f, 0.f, 0.f};
      c = __builtin_amdgcn_wmma_f32_16x16x32_bf16(false, a[0].v, false, b0,
                                                  (short)0, c, false, false);
      c = __builtin_amdgcn_wmma_f32_16x16x32_bf16(false, a[1].v, false, b1,
                                                  (short)0, c, false, false);
      c = __builtin_amdgcn_wmma_f32_16x16x32_bf16(false, a[2].v, false, b2,
                                                  (short)0, c, false, false);
      c = __builtin_amdgcn_wmma_f32_16x16x32_bf16(false, a[3].v, false, b3,
                                                  (short)0, c, false, false);
#pragma unroll
      for (int v = 0; v < 8; ++v) acc[v] += __expf(c[v]);
    }
    __syncthreads();                 // all waves done reading buf before reuse
  }

  // reduce across the 16 lanes of each half (columns) -> row partial sums
#pragma unroll
  for (int off = 1; off < 16; off <<= 1) {
#pragma unroll
    for (int v = 0; v < 8; ++v) acc[v] += __shfl_xor(acc[v], off, 32);
  }
  if (mrow == 0) {  // lanes 0/16 hold rows m0..m0+7 / m0+8..m0+15
#pragma unroll
    for (int v = 0; v < 8; ++v) atomicAdd(&norms[m0 + half * 8 + v], acc[v]);
  }
}

// ---------------- visit_rep[b][d] = tanh(sum_l masked E[x[b,l]][d]) ---------
__global__ void __launch_bounds__(128)
visit_rep_k(const int* __restrict__ x, const float* __restrict__ E,
            float* __restrict__ vr) {
  const int b = blockIdx.x;
  const int d = threadIdx.x;
  float s = 0.0f;
#pragma unroll 4
  for (int l = 0; l < L_SZ; ++l) {
    int code = x[b * L_SZ + l];          // uniform -> scalar load
    if (code != 0) s += E[(size_t)code * D_SZ + d];
  }
  vr[b * D_SZ + d] = tanhf(s);
}

// ---------------- logits + softmax + BCE, one block per visit ---------------
__global__ void __launch_bounds__(256)
visit_cost_k(const float* __restrict__ vr, const float* __restrict__ W,
             const float* __restrict__ bias, const float* __restrict__ label,
             float* __restrict__ acc) {
  __shared__ float sv[D_SZ];
  __shared__ float red[256];
  const int b = blockIdx.x;
  const int t = threadIdx.x;

  if (t < D_SZ) sv[t] = vr[b * D_SZ + t];
  __syncthreads();

  float logit[6];
#pragma unroll
  for (int q = 0; q < 6; ++q) {
    int c = t + q * 256;
    float s = -1e30f;
    if (c < C_SZ) {
      s = bias[c];
      for (int d = 0; d < D_SZ; ++d) s += sv[d] * W[d * C_SZ + c];
    }
    logit[q] = s;
  }

  // block max
  float mx = logit[0];
#pragma unroll
  for (int q = 1; q < 6; ++q) mx = fmaxf(mx, logit[q]);
  red[t] = mx;
  __syncthreads();
  for (int s = 128; s > 0; s >>= 1) {
    if (t < s) red[t] = fmaxf(red[t], red[t + s]);
    __syncthreads();
  }
  mx = red[0];
  __syncthreads();

  // block sum of exp
  float es = 0.0f;
#pragma unroll
  for (int q = 0; q < 6; ++q) {
    int c = t + q * 256;
    if (c < C_SZ) es += __expf(logit[q] - mx);
  }
  red[t] = es;
  __syncthreads();
  for (int s = 128; s > 0; s >>= 1) {
    if (t < s) red[t] += red[t + s];
    __syncthreads();
  }
  float Z = red[0];
  __syncthreads();

  // BCE-style cost terms
  float cost = 0.0f;
#pragma unroll
  for (int q = 0; q < 6; ++q) {
    int c = t + q * 256;
    if (c < C_SZ) {
      float p = __expf(logit[q] - mx) / Z;
      float y = label[b * C_SZ + c];
      cost += y * logf(p + 1e-5f) + (1.0f - y) * logf(1.0f - p + 1e-5f);
    }
  }
  red[t] = cost;
  __syncthreads();
  for (int s = 128; s > 0; s >>= 1) {
    if (t < s) red[t] += red[t + s];
    __syncthreads();
  }
  if (t == 0) atomicAdd(acc, red[0]);
}

// ---------------- concept pairs: one wave per pair --------------------------
__global__ void __launch_bounds__(256)
pair_k(const int* __restrict__ iv, const int* __restrict__ jv,
       const float* __restrict__ E, const float* __restrict__ norms,
       float* __restrict__ acc) {
  __shared__ float red[8];
  const int lane = threadIdx.x & 31;
  const int wave = threadIdx.x >> 5;
  const int p = blockIdx.x * 8 + wave;

  float dot = 0.0f;
  int i = 0;
  if (p < P_SZ) {
    i = iv[p];
    int j = jv[p];
    const float4* ar = (const float4*)(E + (size_t)i * D_SZ);
    const float4* br = (const float4*)(E + (size_t)j * D_SZ);
    float4 xa = ar[lane];
    float4 xb = br[lane];
    dot = xa.x * xb.x + xa.y * xb.y + xa.z * xb.z + xa.w * xb.w;
  }
#pragma unroll
  for (int off = 16; off > 0; off >>= 1) dot += __shfl_xor(dot, off, 32);

  float term = 0.0f;
  if (lane == 0 && p < P_SZ) {
    term = -logf(__expf(dot) / norms[i] + 1e-8f);
  }
  if (lane == 0) red[wave] = term;
  __syncthreads();
  if (threadIdx.x == 0) {
    float s = 0.0f;
#pragma unroll
    for (int w = 0; w < 8; ++w) s += red[w];
    atomicAdd(acc, s);
  }
}

// ---------------- combine to final scalar -----------------------------------
__global__ void combine_k(const float* __restrict__ accs,
                          float* __restrict__ out) {
  out[0] = -accs[0] / (float)B_SZ + accs[1] / (float)P_SZ;
}

// ---------------------------------------------------------------------------
extern "C" void kernel_launch(void* const* d_in, const int* in_sizes, int n_in,
                              void* d_out, int out_size, void* d_ws,
                              size_t ws_size, hipStream_t stream) {
  (void)in_sizes; (void)n_in; (void)out_size; (void)ws_size;
  const int*   x_batch = (const int*)d_in[0];
  const int*   i_vec   = (const int*)d_in[1];
  const int*   j_vec   = (const int*)d_in[2];
  const float* label   = (const float*)d_in[3];
  const float* emb     = (const float*)d_in[4];
  const float* W_cls   = (const float*)d_in[5];
  const float* b_cls   = (const float*)d_in[6];
  float* out = (float*)d_out;

  char* ws = (char*)d_ws;
  unsigned short* Eb = (unsigned short*)ws;                 // 3,145,728 B
  float* norms = (float*)(ws + 3145728);                    //    49,152 B
  float* vr    = (float*)(ws + 3145728 + 49152);            //   131,072 B
  float* accs  = (float*)(ws + 3145728 + 49152 + 131072);   //         8 B

  init_k<<<(V_SZ + 255) / 256, 256, 0, stream>>>(norms, accs);

  const int n_elems = V_SZ * D_SZ;                          // 1,572,864
  cvt_bf16_k<<<(n_elems + 255) / 256, 256, 0, stream>>>(emb, Eb, n_elems);

  // 192 M-supertiles x 2 N-halves = 384 blocks of 128 threads (4 waves)
  gram_norms_k<<<(V_SZ / GRAM_MROWS) * GRAM_NSPLIT, 128, 0, stream>>>(Eb, norms);

  visit_rep_k<<<B_SZ, D_SZ, 0, stream>>>(x_batch, emb, vr);
  visit_cost_k<<<B_SZ, 256, 0, stream>>>(vr, W_cls, b_cls, label, accs + 0);

  pair_k<<<P_SZ / 8, 256, 0, stream>>>(i_vec, j_vec, emb, norms, accs + 1);

  combine_k<<<1, 1, 0, stream>>>(accs, out);
}
